// LexiNeoTCRModel_80702435492490
// MI455X (gfx1250) — matrix operand order repaired and checked by reading, not compiled, not run on hardware
//
#include <hip/hip_runtime.h>
#include <hip/hip_bf16.h>
#include <math.h>

// ---------------- model constants ----------------
constexpr int NB     = 8192;   // batch
constexpr int NPROTO = 8192;   // prototypes
constexpr int EDIM   = 256;    // embedding dim
constexpr int HD     = 256;    // hidden
constexpr int BNC    = 128;    // BN proj dim
constexpr int PL     = 15;     // peptide len
constexpr int ML     = 34;     // MHC len
constexpr int PADL   = 23;     // PL + 2*4 (pad for kernel 9)
constexpr int CC     = 128;    // conv channels
constexpr int NQL    = 4;      // queries
constexpr int TK     = 32;     // top-k
constexpr int NHD    = 8;      // heads
constexpr int DHD    = 32;     // head dim
constexpr int FEAT   = 384;    // 3*CC

typedef __attribute__((ext_vector_type(16))) _Float16 v16h;
typedef __attribute__((ext_vector_type(8)))  _Float16 v8h;
typedef __attribute__((ext_vector_type(8)))  float    v8f;

__device__ __forceinline__ v16h pack16(v8h lo, v8h hi) {
  v16h f;
#pragma unroll
  for (int i = 0; i < 8; ++i) { f[i] = lo[i]; f[i + 8] = hi[i]; }
  return f;
}

// Read all four 32x16 B fragments from the LDS tile, then issue the four
// WMMAs back-to-back (independent accumulators -> no inter-WMMA hazards).
__device__ __forceinline__ void wmma_step(const _Float16* lbase, int lrowbase, int lkoff,
                                          v16h a, v8f acc[4]) {
  v16h bf[4];
#pragma unroll
  for (int t = 0; t < 4; ++t) {
    const _Float16* lp = lbase + (lrowbase + t * 16) * 32 + lkoff;
    bf[t] = pack16(*(const v8h*)lp, *(const v8h*)(lp + 16));
  }
#pragma unroll
  for (int t = 0; t < 4; ++t)
    acc[t] = __builtin_amdgcn_wmma_f32_16x16x32_f16(
        false, a, false, bf[t], (short)0, acc[t], false, false);
}

__device__ __forceinline__ float block_reduce_sum(float v, float* sbuf) {
  int t = threadIdx.x;
  sbuf[t] = v; __syncthreads();
  for (int s = blockDim.x >> 1; s > 0; s >>= 1) {
    if (t < s) sbuf[t] += sbuf[t + s];
    __syncthreads();
  }
  float r = sbuf[0]; __syncthreads();
  return r;
}

__device__ __forceinline__ float sigmoidf_(float x) { return 1.f / (1.f + expf(-x)); }

// =================================================================
// Generic WMMA GEMM:  C[M,N] = A[M,K] @ Bt[N,K]^T   (A,Bt f16 row-major)
// block = 256 threads = 8 waves; block tile 64x128; wave tile 16x64.
// B tile (128 rows x 32 halfs) staged in LDS, double-buffered; the
// steady-state loop body is branch-free: next-step global loads are
// issued before the current step's WMMAs; last step peeled.
// grid = dim3(N/128, M/64).
// =================================================================
template <bool HALF_OUT, bool ACCUM>
__global__ __launch_bounds__(256) void gemm_wmma_kernel(
    const _Float16* __restrict__ A, int lda,
    const _Float16* __restrict__ Bt, int ldb,
    void* __restrict__ Cv, int M, int N, int K) {
  __shared__ _Float16 sB[2][128 * 32];   // [buf][row*32 + khalf]
  const int tid  = threadIdx.x;
  const int lane = tid & 31;
  const int wave = tid >> 5;
  const int row0 = blockIdx.y * 64 + (wave >> 1) * 16;
  const int colB = blockIdx.x * 128;           // block col base
  const int col0 = colB + (wave & 1) * 64;     // wave col base
  (void)M;
  // ---- cooperative B-tile fill mapping: 512 chunks of 8 halfs ----
  const int frow = tid >> 2;                   // 0..63
  const int fpart = (tid & 3) * 8;             // half offset within 32
  const _Float16* gB0 = Bt + (size_t)(colB + frow) * ldb + fpart;
  const _Float16* gB1 = gB0 + (size_t)64 * ldb;
  const int l0 = frow * 32 + fpart;
  const int l1 = l0 + 64 * 32;
  // ---- A fragment pointer (16x32 f16, ISA 7.12.2 layout) ----
  const _Float16* Arow = A + (size_t)(row0 + (lane & 15)) * lda + ((lane >> 4) << 3);
  // ---- B fragment source in LDS ----
  const int lrowbase = (wave & 1) * 64 + (lane & 15);
  const int lkoff    = (lane >> 4) << 3;
  const int lr = ((lane >> 4) << 3);
  const int cn = lane & 15;

  v8f acc[4] = {};
  if (ACCUM) {
    const float* Cf = (const float*)Cv;
#pragma unroll
    for (int t = 0; t < 4; ++t)
#pragma unroll
      for (int v = 0; v < 8; ++v)
        acc[t][v] = Cf[(size_t)(row0 + lr + v) * N + col0 + t * 16 + cn];
  }

  // preload k-step 0 into LDS buffer 0 + A fragment into registers
  {
    v8h r0 = *(const v8h*)(gB0);
    v8h r1 = *(const v8h*)(gB1);
    *(v8h*)(&sB[0][l0]) = r0;
    *(v8h*)(&sB[0][l1]) = r1;
  }
  v16h a_cur = pack16(*(const v8h*)(Arow), *(const v8h*)(Arow + 16));
  __syncthreads();

  int buf = 0;
  int k0 = 0;
  for (; k0 + 32 < K; k0 += 32) {            // steady state (branch-free body)
    const int kn = k0 + 32;
    __builtin_prefetch(Arow + kn + 256, 0, 1);     // global_prefetch_b8
    v8h  n0 = *(const v8h*)(gB0 + kn);             // next-step loads in flight
    v8h  n1 = *(const v8h*)(gB1 + kn);
    v16h a_nxt = pack16(*(const v8h*)(Arow + kn), *(const v8h*)(Arow + kn + 16));
    wmma_step(&sB[buf][0], lrowbase, lkoff, a_cur, acc);
    *(v8h*)(&sB[buf ^ 1][l0]) = n0;
    *(v8h*)(&sB[buf ^ 1][l1]) = n1;
    __syncthreads();                         // s_wait_dscnt + s_barrier
    a_cur = a_nxt;
    buf ^= 1;
  }
  wmma_step(&sB[buf][0], lrowbase, lkoff, a_cur, acc);   // peeled last step

  if (HALF_OUT) {
    _Float16* Ch = (_Float16*)Cv;
#pragma unroll
    for (int t = 0; t < 4; ++t)
#pragma unroll
      for (int v = 0; v < 8; ++v)
        Ch[(size_t)(row0 + lr + v) * N + col0 + t * 16 + cn] = (_Float16)acc[t][v];
  } else {
    float* Cf = (float*)Cv;
#pragma unroll
    for (int t = 0; t < 4; ++t)
#pragma unroll
      for (int v = 0; v < 8; ++v)
        Cf[(size_t)(row0 + lr + v) * N + col0 + t * 16 + cn] = acc[t][v];
  }
}

// =================================================================
// Conv-as-tap-GEMM with LDS double-buffered weight tiles:
//   y[B*15, 128] = sum_j xpad[row(m)+j] @ Wj^T
// xpad: [B, 23, 256] f16 zero-padded;  wt: [KS][128][256] f16.
// Virtual-K loop of KS*8 steps over (tap j, k-block); branch-free
// steady state + peeled last step, same as gemm_wmma_kernel.
// grid = dim3(1, (B*15)/64)
// =================================================================
__global__ __launch_bounds__(256) void conv_gemm_kernel(
    const _Float16* __restrict__ xpad, const _Float16* __restrict__ wt,
    float* __restrict__ y, int KS) {
  __shared__ _Float16 sB[2][128 * 32];
  const int tid  = threadIdx.x;
  const int lane = tid & 31;
  const int wave = tid >> 5;
  const int row0 = blockIdx.y * 64 + (wave >> 1) * 16;
  const int col0 = (wave & 1) * 64;
  // per-lane A row mapping (m -> padded input row)
  const int m  = row0 + (lane & 15);
  const int bb = m / PL;
  const int tt = m - bb * PL;
  const _Float16* Abase =
      xpad + (size_t)(bb * PADL + tt + 4 - (KS >> 1)) * EDIM + ((lane >> 4) << 3);
  // cooperative weight-tile fill mapping
  const int frow  = tid >> 2;            // 0..63
  const int fpart = (tid & 3) * 8;       // half offset within 32
  const int l0 = frow * 32 + fpart;
  const int l1 = l0 + 64 * 32;
  // LDS fragment read mapping
  const int lrowbase = (wave & 1) * 64 + (lane & 15);
  const int lkoff    = (lane >> 4) << 3;

  const int nsteps = KS * 8;             // (tap, k-block) steps, 32 halfs each
  // preload step 0
  {
    const _Float16* g = wt + (size_t)frow * EDIM + fpart;   // j=0, k0=0
    *(v8h*)(&sB[0][l0]) = *(const v8h*)g;
    *(v8h*)(&sB[0][l1]) = *(const v8h*)(g + (size_t)64 * EDIM);
  }
  v16h a_cur = pack16(*(const v8h*)(Abase), *(const v8h*)(Abase + 16));
  __syncthreads();

  v8f acc[4] = {};
  int buf = 0;
  for (int s = 0; s + 1 < nsteps; ++s) {   // steady state (branch-free body)
    const int jn  = (s + 1) >> 3;
    const int k0n = ((s + 1) & 7) << 5;
    const _Float16* g = wt + (size_t)jn * CC * EDIM + (size_t)frow * EDIM + k0n + fpart;
    v8h n0 = *(const v8h*)g;
    v8h n1 = *(const v8h*)(g + (size_t)64 * EDIM);
    const _Float16* ap = Abase + (size_t)jn * EDIM + k0n;
    v16h a_nxt = pack16(*(const v8h*)ap, *(const v8h*)(ap + 16));
    wmma_step(&sB[buf][0], lrowbase, lkoff, a_cur, acc);
    *(v8h*)(&sB[buf ^ 1][l0]) = n0;
    *(v8h*)(&sB[buf ^ 1][l1]) = n1;
    __syncthreads();
    a_cur = a_nxt;
    buf ^= 1;
  }
  wmma_step(&sB[buf][0], lrowbase, lkoff, a_cur, acc);   // peeled last step

  const int lr = ((lane >> 4) << 3);
  const int cn = lane & 15;
#pragma unroll
  for (int t = 0; t < 4; ++t)
#pragma unroll
    for (int v = 0; v < 8; ++v)
      y[(size_t)(row0 + lr + v) * CC + col0 + t * 16 + cn] = acc[t][v];
}

// ---------------- weight prep ----------------
// out[n*K + k] = in[k*ldin + n]   (f32 -> f16 transpose)
__global__ void transpose_f16_kernel(const float* __restrict__ in, _Float16* __restrict__ out,
                                     int K, int N, int ldin) {
  for (long long i = blockIdx.x * (long long)blockDim.x + threadIdx.x;
       i < (long long)K * N; i += (long long)gridDim.x * blockDim.x) {
    int k = (int)(i / N), n = (int)(i % N);
    out[(size_t)n * K + k] = (_Float16)in[(size_t)k * ldin + n];
  }
}

// conv_w [128][256][KS] f32 -> wt [KS][128][256] f16
__global__ void convw_prep_kernel(const float* __restrict__ w, _Float16* __restrict__ out, int KS) {
  int total = CC * EDIM * KS;
  for (int i = blockIdx.x * blockDim.x + threadIdx.x; i < total; i += gridDim.x * blockDim.x) {
    int o = i / (EDIM * KS); int rem = i % (EDIM * KS);
    int c = rem / KS; int j = rem % KS;
    out[(size_t)j * CC * EDIM + (size_t)o * EDIM + c] = (_Float16)w[i];
  }
}

// prototypes f32 [K,256] -> raw f16 + L2-normalized f16
__global__ __launch_bounds__(256) void proto_prep_kernel(
    const float* __restrict__ protos, _Float16* __restrict__ raw, _Float16* __restrict__ nrm) {
  int k = blockIdx.x, h = threadIdx.x;
  float v = protos[(size_t)k * HD + h];
  raw[(size_t)k * HD + h] = (_Float16)v;
  __shared__ float sbuf[256];
  float ss = block_reduce_sum(v * v, sbuf);
  float inv = 1.f / fmaxf(sqrtf(ss), 1e-12f);
  nrm[(size_t)k * HD + h] = (_Float16)(v * inv);
}

// ---------------- embeddings ----------------
__global__ __launch_bounds__(256) void embed_kernel(
    const int* __restrict__ pep_tok, const int* __restrict__ mhc_tok,
    const float* __restrict__ pep_e, const float* __restrict__ mhc_e,
    const float* __restrict__ pos_e,
    _Float16* __restrict__ pep16, _Float16* __restrict__ mhc16, int* __restrict__ lens) {
  int b = blockIdx.x, tid = threadIdx.x;
  if (tid == 0) {
    int c = 0;
    for (int i = 0; i < PL; ++i) c += (pep_tok[(size_t)b * PL + i] != 0);
    lens[b] = c;
  }
  for (int e = tid; e < PL * EDIM; e += 256) {
    int pos = e / EDIM, c = e % EDIM;
    int tok = pep_tok[(size_t)b * PL + pos];
    float v = (tok != 0) ? pep_e[(size_t)tok * EDIM + c] : 0.f;
    pep16[(size_t)b * PL * EDIM + e] = (_Float16)v;
  }
  for (int e = tid; e < ML * EDIM; e += 256) {
    int pos = e / EDIM, c = e % EDIM;
    int tok = mhc_tok[(size_t)b * ML + pos];
    mhc16[(size_t)b * ML * EDIM + e] =
        (_Float16)(mhc_e[(size_t)tok * EDIM + c] + pos_e[(size_t)pos * EDIM + c]);
  }
}

// bias + LayerNorm over 128 -> f16 (one row per block, blockDim=128)
__global__ __launch_bounds__(128) void ln128_kernel(
    const float* __restrict__ pre, const float* __restrict__ bias,
    const float* __restrict__ g, const float* __restrict__ beta, _Float16* __restrict__ out) {
  int r = blockIdx.x, c = threadIdx.x;
  float x = pre[(size_t)r * BNC + c] + bias[c];
  __shared__ float sbuf[128];
  float mean = block_reduce_sum(x, sbuf) * (1.f / BNC);
  float d = x - mean;
  float var = block_reduce_sum(d * d, sbuf) * (1.f / BNC);
  out[(size_t)r * BNC + c] = (_Float16)(d * rsqrtf(var + 1e-5f) * g[c] + beta[c]);
}

// ---------------- pep/MHC interaction (one block per batch) ----------------
__global__ __launch_bounds__(256) void interaction_kernel(
    const _Float16* __restrict__ pepbn, const _Float16* __restrict__ mhcbn,
    const _Float16* __restrict__ pepemb, const _Float16* __restrict__ mhcemb,
    _Float16* __restrict__ xpad) {
  int b = blockIdx.x, tid = threadIdx.x;
  __shared__ _Float16 sPbn[PL * BNC];
  __shared__ _Float16 sMbn[ML * BNC];
  __shared__ _Float16 sMemb[ML * EDIM];
  __shared__ _Float16 sPemb[PL * EDIM];
  __shared__ float    sC[PL * ML];
  __shared__ float    sGate[PL];
  for (int i = tid; i < PL * BNC; i += 256)  sPbn[i]  = pepbn[(size_t)b * PL * BNC + i];
  for (int i = tid; i < ML * BNC; i += 256)  sMbn[i]  = mhcbn[(size_t)b * ML * BNC + i];
  for (int i = tid; i < ML * EDIM; i += 256) sMemb[i] = mhcemb[(size_t)b * ML * EDIM + i];
  for (int i = tid; i < PL * EDIM; i += 256) sPemb[i] = pepemb[(size_t)b * PL * EDIM + i];
  __syncthreads();
  for (int e = tid; e < PL * ML; e += 256) {
    int pp = e / ML, mm = e % ML;
    float s = 0.f;
    for (int c = 0; c < BNC; ++c) s += (float)sPbn[pp * BNC + c] * (float)sMbn[mm * BNC + c];
    sC[e] = s * 0.08838834764831845f;  // 1/sqrt(128)
  }
  __syncthreads();
  if (tid < PL) {
    float mx = -1e30f;
    for (int mm = 0; mm < ML; ++mm) mx = fmaxf(mx, sC[tid * ML + mm]);
    sGate[tid] = sigmoidf_(mx);
    float ssum = 0.f;
    for (int mm = 0; mm < ML; ++mm) { float e2 = expf(sC[tid * ML + mm] - mx); sC[tid * ML + mm] = e2; ssum += e2; }
    float inv = 1.f / ssum;
    for (int mm = 0; mm < ML; ++mm) sC[tid * ML + mm] *= inv;
  }
  __syncthreads();
  _Float16* xb = xpad + (size_t)b * PADL * EDIM;
  for (int i = tid; i < 4 * EDIM; i += 256) {            // zero pad rows (4 front, 4 back)
    xb[i] = (_Float16)0.f;
    xb[(size_t)(PADL - 4) * EDIM + i] = (_Float16)0.f;
  }
  for (int e = tid; e < PL * EDIM; e += 256) {
    int pp = e / EDIM, c = e % EDIM;
    float s = 0.f;
    for (int mm = 0; mm < ML; ++mm) s += sC[pp * ML + mm] * (float)sMemb[mm * EDIM + c];
    float v = (float)sPemb[e] * s * sGate[pp];
    xb[(size_t)(pp + 4) * EDIM + c] = (_Float16)v;
  }
}

// ---------------- conv BN+ReLU+maxpool -> feats slice ----------------
__global__ void pool_bn_relu_max_kernel(
    const float* __restrict__ y, const float* __restrict__ cb,
    const float* __restrict__ g, const float* __restrict__ bb,
    const float* __restrict__ mean, const float* __restrict__ var,
    _Float16* __restrict__ feats, int off) {
  int idx = blockIdx.x * blockDim.x + threadIdx.x;
  if (idx >= NB * CC) return;
  int b = idx / CC, c = idx % CC;
  float inv = rsqrtf(var[c] + 1e-5f);
  float gc = g[c], bbc = bb[c], cbc = cb[c], mc = mean[c];
  const float* yb = y + (size_t)b * PL * CC + c;
  float best = 0.f;
  for (int t = 0; t < PL; ++t) {
    float v = (yb[(size_t)t * CC] + cbc - mc) * inv * gc + bbc;
    best = fmaxf(best, fmaxf(v, 0.f));
  }
  feats[(size_t)b * FEAT + off + c] = (_Float16)best;
}

// ---------------- antigen epilog: bias + LN + len-emb; init context ----------------
__global__ __launch_bounds__(256) void antigen_epilog_kernel(
    const float* __restrict__ projpre, const float* __restrict__ proj_b,
    const float* __restrict__ lng, const float* __restrict__ lnb,
    const float* __restrict__ lenemb, const int* __restrict__ lens,
    const float* __restrict__ qemb, float* __restrict__ antigen,
    float* __restrict__ ctxf, _Float16* __restrict__ ctx16) {
  int b = blockIdx.x, h = threadIdx.x;
  float x = projpre[(size_t)b * HD + h] + proj_b[h];
  __shared__ float sbuf[256];
  float mean = block_reduce_sum(x, sbuf) * (1.f / HD);
  float d = x - mean;
  float var = block_reduce_sum(d * d, sbuf) * (1.f / HD);
  float y = d * rsqrtf(var + 1e-5f) * lng[h] + lnb[h];
  int L = lens[b]; if (L > PL) L = PL;
  y += lenemb[(size_t)L * HD + h];
  antigen[(size_t)b * HD + h] = y;
  for (int q = 0; q < NQL; ++q) {
    float c = qemb[(size_t)q * HD + h] + y;
    ctxf[((size_t)b * NQL + q) * HD + h] = c;
    ctx16[((size_t)b * NQL + q) * HD + h] = (_Float16)c;
  }
}

// ---------------- context mean + L2 normalize -> f16 ----------------
__global__ __launch_bounds__(256) void ctx_norm_kernel(
    const float* __restrict__ ctx, _Float16* __restrict__ out) {
  int b = blockIdx.x, h = threadIdx.x;
  const float* cb = ctx + (size_t)b * NQL * HD;
  float m = 0.25f * (cb[h] + cb[HD + h] + cb[2 * HD + h] + cb[3 * HD + h]);
  __shared__ float sbuf[256];
  float ss = block_reduce_sum(m * m, sbuf);
  float inv = 1.f / fmaxf(sqrtf(ss), 1e-12f);
  out[(size_t)b * HD + h] = (_Float16)(m * inv);
}

// ---------------- top-32 of 8192 + gather prototypes ----------------
__global__ __launch_bounds__(256) void topk_gather_kernel(
    const float* __restrict__ sims, const _Float16* __restrict__ proto16,
    _Float16* __restrict__ sel) {
  int b = blockIdx.x, t = threadIdx.x;
  const float* srow = sims + (size_t)b * NPROTO;
  float vals[32];
#pragma unroll
  for (int i = 0; i < 32; ++i) vals[i] = srow[i * 256 + t];
  unsigned used = 0;
  __shared__ float sv[256];
  __shared__ int   si[256];
  __shared__ int   widx[TK];
  for (int r = 0; r < TK; ++r) {
    float best = -3.4e38f; int bi = 0;
#pragma unroll
    for (int i = 0; i < 32; ++i)
      if (!(used & (1u << i)) && vals[i] > best) { best = vals[i]; bi = i; }
    sv[t] = best; si[t] = bi * 256 + t;
    __syncthreads();
    for (int s2 = 128; s2 > 0; s2 >>= 1) {
      if (t < s2 && sv[t + s2] > sv[t]) { sv[t] = sv[t + s2]; si[t] = si[t + s2]; }
      __syncthreads();
    }
    int win = si[0];
    if (t == 0) widx[r] = win;
    if ((win & 255) == t) used |= (1u << (win >> 8));
    __syncthreads();
  }
  for (int c = t; c < TK * (HD / 8); c += 256) {
    int r = c / (HD / 8), c8 = (c % (HD / 8)) * 8;
    int pidx = widx[r];
    *(v8h*)(sel + ((size_t)b * TK + r) * HD + c8) =
        *(const v8h*)(proto16 + (size_t)pidx * HD + c8);
  }
}

// ---------------- MHA core (per batch): softmax(QK^T/sqrt(dh)) V ----------------
__global__ __launch_bounds__(256) void attn_core_kernel(
    const _Float16* __restrict__ q16, const _Float16* __restrict__ k16,
    const _Float16* __restrict__ v16p, const float* __restrict__ ipb,
    _Float16* __restrict__ attno16) {
  int b = blockIdx.x, tid = threadIdx.x;
  __shared__ float sQ[NQL * HD];
  __shared__ float sK[TK * HD];
  __shared__ float sS[NHD * NQL * TK];
  for (int i = tid; i < NQL * HD; i += 256)
    sQ[i] = (float)q16[(size_t)b * NQL * HD + i] + ipb[i % HD];
  for (int i = tid; i < TK * HD; i += 256)
    sK[i] = (float)k16[(size_t)b * TK * HD + i] + ipb[HD + i % HD];
  __syncthreads();
  for (int e = tid; e < NHD * NQL * TK; e += 256) {
    int hh = e / (NQL * TK), qq = (e / TK) % NQL, kk = e % TK;
    float s = 0.f;
    for (int d = 0; d < DHD; ++d) s += sQ[qq * HD + hh * DHD + d] * sK[kk * HD + hh * DHD + d];
    sS[e] = s * 0.17677669529663687f;  // 1/sqrt(32)
  }
  __syncthreads();
  if (tid < NHD * NQL) {
    float mx = -1e30f;
    for (int kk = 0; kk < TK; ++kk) mx = fmaxf(mx, sS[tid * TK + kk]);
    float ss = 0.f;
    for (int kk = 0; kk < TK; ++kk) { float e2 = expf(sS[tid * TK + kk] - mx); sS[tid * TK + kk] = e2; ss += e2; }
    float inv = 1.f / ss;
    for (int kk = 0; kk < TK; ++kk) sS[tid * TK + kk] *= inv;
  }
  __syncthreads();
  const _Float16* vb = v16p + (size_t)b * TK * HD;
  for (int e = tid; e < NQL * HD; e += 256) {
    int qq = e / HD, c = e % HD, hh = c / DHD;
    float bvc = ipb[2 * HD + c];
    float o = 0.f;
    for (int kk = 0; kk < TK; ++kk)
      o += sS[(hh * NQL + qq) * TK + kk] * ((float)vb[(size_t)kk * HD + c] + bvc);
    attno16[(size_t)b * NQL * HD + e] = (_Float16)o;
  }
}

// out-proj bias; produce f32 + f16 copies
__global__ void attn_epilog_kernel(const float* __restrict__ pre, const float* __restrict__ ob,
                                   float* __restrict__ attnf, _Float16* __restrict__ attn16) {
  long long n = (long long)NB * NQL * HD;
  for (long long i = blockIdx.x * (long long)blockDim.x + threadIdx.x; i < n;
       i += (long long)gridDim.x * blockDim.x) {
    float v = pre[i] + ob[i % HD];
    attnf[i] = v;
    attn16[i] = (_Float16)v;
  }
}

// context += sigmoid(gpre + gb) * attn
__global__ void update_ctx_kernel(const float* __restrict__ gpre, const float* __restrict__ gb,
                                  const float* __restrict__ attnf,
                                  float* __restrict__ ctxf, _Float16* __restrict__ ctx16) {
  long long n = (long long)NB * NQL * HD;
  for (long long i = blockIdx.x * (long long)blockDim.x + threadIdx.x; i < n;
       i += (long long)gridDim.x * blockDim.x) {
    float g = sigmoidf_(gpre[i] + gb[i % HD]);
    float c = ctxf[i] + g * attnf[i];
    ctxf[i] = c;
    ctx16[i] = (_Float16)c;
  }
}

// ---------------- final: combine + LN + scorer + sigmoid ----------------
__global__ __launch_bounds__(256) void final_kernel(
    const float* __restrict__ antigen, const float* __restrict__ pooled,
    const float* __restrict__ pool_b, const float* __restrict__ scale,
    const float* __restrict__ lng, const float* __restrict__ lnb,
    const float* __restrict__ sw, const float* __restrict__ sb, float* __restrict__ out) {
  int b = blockIdx.x, h = threadIdx.x;
  float x = antigen[(size_t)b * HD + h] + scale[0] * (pooled[(size_t)b * HD + h] + pool_b[h]);
  __shared__ float sbuf[256];
  float mean = block_reduce_sum(x, sbuf) * (1.f / HD);
  float d = x - mean;
  float var = block_reduce_sum(d * d, sbuf) * (1.f / HD);
  float y = d * rsqrtf(var + 1e-5f) * lng[h] + lnb[h];
  float dot = block_reduce_sum(y * sw[h], sbuf);
  if (h == 0) out[b] = sigmoidf_(dot + sb[0]);
}

// =================================================================
extern "C" void kernel_launch(void* const* d_in, const int* in_sizes, int n_in,
                              void* d_out, int out_size, void* d_ws, size_t ws_size,
                              hipStream_t stream) {
  (void)in_sizes; (void)n_in; (void)out_size; (void)ws_size;
  // ---- inputs (setup_inputs dict order) ----
  const int*   pep_tok   = (const int*)d_in[0];
  const int*   mhc_tok   = (const int*)d_in[1];
  const float* pep_embed = (const float*)d_in[2];
  const float* mhc_embed = (const float*)d_in[3];
  const float* pos_embed = (const float*)d_in[4];
  const float* bn_pep_w  = (const float*)d_in[5];
  const float* bn_pep_b  = (const float*)d_in[6];
  const float* bn_pep_g  = (const float*)d_in[7];
  const float* bn_pep_be = (const float*)d_in[8];
  const float* bn_mhc_w  = (const float*)d_in[9];
  const float* bn_mhc_b  = (const float*)d_in[10];
  const float* bn_mhc_g  = (const float*)d_in[11];
  const float* bn_mhc_be = (const float*)d_in[12];
  const float* conv_w[3]  = {(const float*)d_in[13], (const float*)d_in[19], (const float*)d_in[25]};
  const float* conv_b[3]  = {(const float*)d_in[14], (const float*)d_in[20], (const float*)d_in[26]};
  const float* bnk_g[3]   = {(const float*)d_in[15], (const float*)d_in[21], (const float*)d_in[27]};
  const float* bnk_b[3]   = {(const float*)d_in[16], (const float*)d_in[22], (const float*)d_in[28]};
  const float* bnk_m[3]   = {(const float*)d_in[17], (const float*)d_in[23], (const float*)d_in[29]};
  const float* bnk_v[3]   = {(const float*)d_in[18], (const float*)d_in[24], (const float*)d_in[30]};
  const float* proj_w    = (const float*)d_in[31];
  const float* proj_b    = (const float*)d_in[32];
  const float* cnn_g     = (const float*)d_in[33];
  const float* cnn_b     = (const float*)d_in[34];
  const float* lenemb    = (const float*)d_in[35];
  const float* protos    = (const float*)d_in[36];
  const float* qemb      = (const float*)d_in[37];
  const float* pscale    = (const float*)d_in[38];
  const float* ipw[2]    = {(const float*)d_in[39], (const float*)d_in[45]};
  const float* ipb[2]    = {(const float*)d_in[40], (const float*)d_in[46]};
  const float* ow[2]     = {(const float*)d_in[41], (const float*)d_in[47]};
  const float* ob[2]     = {(const float*)d_in[42], (const float*)d_in[48]};
  const float* gw[2]     = {(const float*)d_in[43], (const float*)d_in[49]};
  const float* gb[2]     = {(const float*)d_in[44], (const float*)d_in[50]};
  const float* pool_w    = (const float*)d_in[51];
  const float* pool_b    = (const float*)d_in[52];
  const float* sln_g     = (const float*)d_in[53];
  const float* sln_b     = (const float*)d_in[54];
  const float* sc_w      = (const float*)d_in[55];
  const float* sc_b      = (const float*)d_in[56];
  float* outF = (float*)d_out;

  // ---- workspace layout ----
  char* wsb = (char*)d_ws;
  size_t off = 0;
  auto take = [&](size_t bytes) -> void* {
    void* p = wsb + off;
    off += (bytes + 255) & ~(size_t)255;
    return p;
  };
  _Float16* wt_bnp  = (_Float16*)take((size_t)BNC * EDIM * 2);
  _Float16* wt_bnm  = (_Float16*)take((size_t)BNC * EDIM * 2);
  _Float16* wt_cv[3];
  const int KS[3] = {5, 7, 9};
  for (int i = 0; i < 3; ++i) wt_cv[i] = (_Float16*)take((size_t)KS[i] * CC * EDIM * 2);
  _Float16* wt_proj = (_Float16*)take((size_t)HD * FEAT * 2);
  _Float16 *wt_q[2], *wt_k[2], *wt_v[2], *wt_o[2], *wt_g[2];
  for (int i = 0; i < 2; ++i) {
    wt_q[i] = (_Float16*)take((size_t)HD * HD * 2);
    wt_k[i] = (_Float16*)take((size_t)HD * HD * 2);
    wt_v[i] = (_Float16*)take((size_t)HD * HD * 2);
    wt_o[i] = (_Float16*)take((size_t)HD * HD * 2);
    wt_g[i] = (_Float16*)take((size_t)HD * 2 * HD * 2);
  }
  _Float16* wt_pool   = (_Float16*)take((size_t)HD * NQL * HD * 2);
  _Float16* proto_raw = (_Float16*)take((size_t)NPROTO * HD * 2);
  _Float16* proto_nrm = (_Float16*)take((size_t)NPROTO * HD * 2);
  _Float16* pep16     = (_Float16*)take((size_t)NB * PL * EDIM * 2);
  _Float16* mhc16     = (_Float16*)take((size_t)NB * ML * EDIM * 2);
  int*      lens      = (int*)take((size_t)NB * 4);
  float*    pre_big   = (float*)take((size_t)NB * ML * BNC * 4);  // reused scratch (largest)
  _Float16* pep_bn16  = (_Float16*)take((size_t)NB * PL * BNC * 2);
  _Float16* mhc_bn16  = (_Float16*)take((size_t)NB * ML * BNC * 2);
  _Float16* xpad      = (_Float16*)take((size_t)NB * PADL * EDIM * 2);
  _Float16* feats16   = (_Float16*)take((size_t)NB * FEAT * 2);
  float*    antigen   = (float*)take((size_t)NB * HD * 4);
  float*    ctxf      = (float*)take((size_t)NB * NQL * HD * 4);
  _Float16* ctx16     = (_Float16*)take((size_t)NB * NQL * HD * 2);
  _Float16* ctxn16    = (_Float16*)take((size_t)NB * HD * 2);
  float*    sims      = (float*)take((size_t)NB * NPROTO * 4);
  _Float16* sel16     = (_Float16*)take((size_t)NB * TK * HD * 2);
  _Float16* q16       = (_Float16*)take((size_t)NB * NQL * HD * 2);
  _Float16* k16       = (_Float16*)take((size_t)NB * TK * HD * 2);
  _Float16* vv16      = (_Float16*)take((size_t)NB * TK * HD * 2);
  _Float16* attno16   = (_Float16*)take((size_t)NB * NQL * HD * 2);
  float*    attnf     = (float*)take((size_t)NB * NQL * HD * 4);
  _Float16* attn16    = (_Float16*)take((size_t)NB * NQL * HD * 2);
  float*    pooled    = (float*)take((size_t)NB * HD * 4);

  // ---- weight prep ----
  transpose_f16_kernel<<<256, 256, 0, stream>>>(bn_pep_w, wt_bnp, EDIM, BNC, BNC);
  transpose_f16_kernel<<<256, 256, 0, stream>>>(bn_mhc_w, wt_bnm, EDIM, BNC, BNC);
  for (int i = 0; i < 3; ++i)
    convw_prep_kernel<<<256, 256, 0, stream>>>(conv_w[i], wt_cv[i], KS[i]);
  transpose_f16_kernel<<<256, 256, 0, stream>>>(proj_w, wt_proj, FEAT, HD, HD);
  for (int i = 0; i < 2; ++i) {
    transpose_f16_kernel<<<256, 256, 0, stream>>>(ipw[i],            wt_q[i], HD, HD, 3 * HD);
    transpose_f16_kernel<<<256, 256, 0, stream>>>(ipw[i] + HD,       wt_k[i], HD, HD, 3 * HD);
    transpose_f16_kernel<<<256, 256, 0, stream>>>(ipw[i] + 2 * HD,   wt_v[i], HD, HD, 3 * HD);
    transpose_f16_kernel<<<256, 256, 0, stream>>>(ow[i],             wt_o[i], HD, HD, HD);
    transpose_f16_kernel<<<256, 256, 0, stream>>>(gw[i],             wt_g[i], 2 * HD, HD, HD);
  }
  transpose_f16_kernel<<<256, 256, 0, stream>>>(pool_w, wt_pool, NQL * HD, HD, HD);
  proto_prep_kernel<<<NPROTO, 256, 0, stream>>>(protos, proto_raw, proto_nrm);

  // ---- embeddings ----
  embed_kernel<<<NB, 256, 0, stream>>>(pep_tok, mhc_tok, pep_embed, mhc_embed, pos_embed,
                                       pep16, mhc16, lens);

  // ---- BN projections + LN ----
  gemm_wmma_kernel<false, false><<<dim3(1, NB * PL / 64), 256, 0, stream>>>(
      pep16, EDIM, wt_bnp, EDIM, pre_big, NB * PL, BNC, EDIM);
  ln128_kernel<<<NB * PL, 128, 0, stream>>>(pre_big, bn_pep_b, bn_pep_g, bn_pep_be, pep_bn16);
  gemm_wmma_kernel<false, false><<<dim3(1, NB * ML / 64), 256, 0, stream>>>(
      mhc16, EDIM, wt_bnm, EDIM, pre_big, NB * ML, BNC, EDIM);
  ln128_kernel<<<NB * ML, 128, 0, stream>>>(pre_big, bn_mhc_b, bn_mhc_g, bn_mhc_be, mhc_bn16);

  // ---- interaction -> padded pMHC ----
  interaction_kernel<<<NB, 256, 0, stream>>>(pep_bn16, mhc_bn16, pep16, mhc16, xpad);

  // ---- convs + BN + ReLU + maxpool -> feats ----
  for (int i = 0; i < 3; ++i) {
    conv_gemm_kernel<<<dim3(1, NB * PL / 64), 256, 0, stream>>>(xpad, wt_cv[i], pre_big, KS[i]);
    pool_bn_relu_max_kernel<<<(NB * CC + 255) / 256, 256, 0, stream>>>(
        pre_big, conv_b[i], bnk_g[i], bnk_b[i], bnk_m[i], bnk_v[i], feats16, i * CC);
  }

  // ---- proj + LN + len-emb -> antigen, init context ----
  gemm_wmma_kernel<false, false><<<dim3(HD / 128, NB / 64), 256, 0, stream>>>(
      feats16, FEAT, wt_proj, FEAT, pre_big, NB, HD, FEAT);
  antigen_epilog_kernel<<<NB, 256, 0, stream>>>(pre_big, proj_b, cnn_g, cnn_b, lenemb, lens,
                                                qemb, antigen, ctxf, ctx16);

  // ---- prototype refinement loop ----
  for (int i = 0; i < 2; ++i) {
    ctx_norm_kernel<<<NB, 256, 0, stream>>>(ctxf, ctxn16);
    gemm_wmma_kernel<false, false><<<dim3(NPROTO / 128, NB / 64), 256, 0, stream>>>(
        ctxn16, HD, proto_nrm, HD, sims, NB, NPROTO, HD);
    topk_gather_kernel<<<NB, 256, 0, stream>>>(sims, proto_raw, sel16);
    gemm_wmma_kernel<true, false><<<dim3(HD / 128, NB * NQL / 64), 256, 0, stream>>>(
        ctx16, HD, wt_q[i], HD, q16, NB * NQL, HD, HD);
    gemm_wmma_kernel<true, false><<<dim3(HD / 128, NB * TK / 64), 256, 0, stream>>>(
        sel16, HD, wt_k[i], HD, k16, NB * TK, HD, HD);
    gemm_wmma_kernel<true, false><<<dim3(HD / 128, NB * TK / 64), 256, 0, stream>>>(
        sel16, HD, wt_v[i], HD, vv16, NB * TK, HD, HD);
    attn_core_kernel<<<NB, 256, 0, stream>>>(q16, k16, vv16, ipb[i], attno16);
    gemm_wmma_kernel<false, false><<<dim3(HD / 128, NB * NQL / 64), 256, 0, stream>>>(
        attno16, HD, wt_o[i], HD, pre_big, NB * NQL, HD, HD);
    attn_epilog_kernel<<<8192, 256, 0, stream>>>(pre_big, ob[i], attnf, attn16);
    gemm_wmma_kernel<false, false><<<dim3(HD / 128, NB * NQL / 64), 256, 0, stream>>>(
        ctx16, HD, wt_g[i], 2 * HD, pre_big, NB * NQL, HD, HD);
    gemm_wmma_kernel<false, true><<<dim3(HD / 128, NB * NQL / 64), 256, 0, stream>>>(
        attn16, HD, wt_g[i] + HD, 2 * HD, pre_big, NB * NQL, HD, HD);
    update_ctx_kernel<<<8192, 256, 0, stream>>>(pre_big, gb[i], attnf, ctxf, ctx16);
  }

  // ---- pool + score ----
  gemm_wmma_kernel<false, false><<<dim3(HD / 128, NB / 64), 256, 0, stream>>>(
      ctx16, NQL * HD, wt_pool, NQL * HD, pooled, NB, HD, NQL * HD);
  final_kernel<<<NB, 256, 0, stream>>>(antigen, pooled, pool_b, pscale, sln_g, sln_b,
                                       sc_w, sc_b, outF);
}
